// MultiheadSelfAttention_42545946034368
// MI455X (gfx1250) — matrix operand order repaired and checked by reading, not compiled, not run on hardware
//
#include <hip/hip_runtime.h>

typedef __attribute__((ext_vector_type(16))) _Float16 v16h;
typedef __attribute__((ext_vector_type(8)))  _Float16 h8;
typedef __attribute__((ext_vector_type(8)))  float    v8f;

constexpr int DMODEL = 1024;
constexpr int NH     = 16;
constexpr int DK     = 64;
constexpr int NB     = 2;          // batch
constexpr int SEQ    = 2048;
constexpr int MROWS  = NB * SEQ;   // 4096
constexpr int QT     = SEQ / 16;   // 128 query tiles per (b,h)

// ---------------------------------------------------------------------------
// Fragment loaders. CDNA5 WMMA f16 layouts (wave32, half = lane>>4):
//   A (16x32): lane -> row M = lane&15; K runs {half*8..+7} and {16+half*8..+7}
//   B (32x16): lane -> col N = lane&15; K runs {half*8..+7} and {16+half*8..+7}
// Both are contiguous f16 runs -> two global_load_b128 per fragment.
// ---------------------------------------------------------------------------
__device__ __forceinline__ v16h load_fragA(const _Float16* p, int hlf) {
  h8 lo = *(const h8*)(p + hlf * 8);
  h8 hi = *(const h8*)(p + 16 + hlf * 8);
  return __builtin_shufflevector(lo, hi, 0,1,2,3,4,5,6,7,8,9,10,11,12,13,14,15);
}
__device__ __forceinline__ v16h load_fragB(const _Float16* p, int hlf) {
  h8 lo = *(const h8*)(p + hlf * 8);
  h8 hi = *(const h8*)(p + 16 + hlf * 8);
  return __builtin_shufflevector(lo, hi, 0,1,2,3,4,5,6,7,8,9,10,11,12,13,14,15);
}
__device__ __forceinline__ v8f wmma16(v16h a, v16h b, v8f c) {
  return __builtin_amdgcn_wmma_f32_16x16x32_f16(false, a, false, b,
                                                (short)0, c, false, false);
}

// ---------------------------------------------------------------------------
// One-pass f32 -> f16 convert (8 elements/thread, b128 in / b128 out).
// ---------------------------------------------------------------------------
__global__ void __launch_bounds__(256)
cvt_kernel(const float* __restrict__ src, _Float16* __restrict__ dst, int n)
{
  int i = (blockIdx.x * 256 + threadIdx.x) * 8;
  if (i >= n) return;
  float4 a = *(const float4*)(src + i);
  float4 b = *(const float4*)(src + i + 4);
  h8 o;
  o[0] = (_Float16)a.x; o[1] = (_Float16)a.y;
  o[2] = (_Float16)a.z; o[3] = (_Float16)a.w;
  o[4] = (_Float16)b.x; o[5] = (_Float16)b.y;
  o[6] = (_Float16)b.z; o[7] = (_Float16)b.w;
  *(h8*)(dst + i) = o;
}

// ---------------------------------------------------------------------------
// GEMM: C[32x64 tile] = A[M,Kd] . W[N,Kd]^T, f16 in, one wave per tile.
// Per K-step: 12 b128 loads feed 8 WMMAs.
// mode 0 -> Y16 [B*H, S, dk] scaled by 1/sqrt(dk) (Q)
// mode 1 -> Y16 [B*H, S, dk]                      (K)
// mode 2 -> Y16 [B*H, dk, S] transposed           (V)
// mode 3 -> Yf  [M, D] f32                        (final output)
// ---------------------------------------------------------------------------
__global__ void __launch_bounds__(32)
gemm_kernel(const _Float16* __restrict__ A, const _Float16* __restrict__ W,
            _Float16* __restrict__ Y16, float* __restrict__ Yf,
            int mode, float scale)
{
  const int lane = threadIdx.x;
  const int hlf  = lane >> 4;
  const int nl   = lane & 15;
  const int row  = blockIdx.x * 32;
  const int colb = blockIdx.y * 64;

  const _Float16* ar0 = A + (size_t)(row + nl) * DMODEL;
  const _Float16* ar1 = A + (size_t)(row + 16 + nl) * DMODEL;

  v8f acc[2][4] = {};
  for (int kk = 0; kk < DMODEL; kk += 32) {
    v16h a0 = load_fragA(ar0 + kk, hlf);
    v16h a1 = load_fragA(ar1 + kk, hlf);
#pragma unroll
    for (int c = 0; c < 4; ++c) {
      const _Float16* wr = W + (size_t)(colb + 16 * c + nl) * DMODEL + kk;
      v16h b = load_fragB(wr, hlf);
      acc[0][c] = wmma16(a0, b, acc[0][c]);
      acc[1][c] = wmma16(a1, b, acc[1][c]);
    }
  }

#pragma unroll
  for (int rt = 0; rt < 2; ++rt) {
#pragma unroll
    for (int c = 0; c < 4; ++c) {
#pragma unroll
      for (int r = 0; r < 8; ++r) {
        int gm = row + rt * 16 + r + 8 * hlf;   // [0, MROWS)
        int gn = colb + 16 * c + nl;            // [0, DMODEL)
        float v = acc[rt][c][r] * scale;
        if (mode == 3) {
          Yf[(size_t)gm * DMODEL + gn] = v;
        } else {
          int bb = gm >> 11;                    // / SEQ
          int ss = gm & (SEQ - 1);
          int hh = gn >> 6;                     // / DK
          int dd = gn & (DK - 1);
          size_t idx = (mode == 2)
              ? ((size_t)(bb * NH + hh) * DK + dd) * SEQ + ss
              : ((size_t)(bb * NH + hh) * SEQ + ss) * DK + dd;
          Y16[idx] = (_Float16)v;
        }
      }
    }
  }
}

// ---------------------------------------------------------------------------
// Flash attention: one wave per (b,h, 16-query tile). Keys streamed in chunks
// of 32; online softmax; O kept as 16x64 f32 in 4 WMMA accumulators.
// Per chunk: 16 global b128 + 2 LDS b128 feed 8 WMMAs.
// ---------------------------------------------------------------------------
__global__ void __launch_bounds__(32)
attn_kernel(const _Float16* __restrict__ Q, const _Float16* __restrict__ K,
            const _Float16* __restrict__ Vt, _Float16* __restrict__ AO)
{
  __shared__ _Float16 ldsP[16 * 32];   // C-layout -> A-layout staging for P

  const int lane  = threadIdx.x;
  const int hlf   = lane >> 4;
  const int nl    = lane & 15;
  const int qt    = blockIdx.x & (QT - 1);
  const int bh    = blockIdx.x >> 7;          // QT == 128
  const int qbase = qt * 16;

  const _Float16* Qp = Q  + (size_t)bh * SEQ * DK;
  const _Float16* Kp = K  + (size_t)bh * SEQ * DK;
  const _Float16* Vp = Vt + (size_t)bh * DK * SEQ;

  // Q A-fragments, dk chunks [0,32) and [32,64). Q pre-scaled by 1/sqrt(dk).
  const _Float16* qr = Qp + (size_t)(qbase + nl) * DK;
  v16h aQ0 = load_fragA(qr, hlf);
  v16h aQ1 = load_fragA(qr + 32, hlf);

  float Mr[8], Lr[8];
  v8f O[4];
#pragma unroll
  for (int r = 0; r < 8; ++r) { Mr[r] = -__builtin_inff(); Lr[r] = 0.0f; }
#pragma unroll
  for (int t = 0; t < 4; ++t) O[t] = (v8f){};

  for (int kb = 0; kb < qbase + 16; kb += 32) {
    // ---- scores S = Q . K^T for a 16x32 key chunk (two 16x16 tiles) ----
    v8f Sv[2];
#pragma unroll
    for (int t = 0; t < 2; ++t) {
      const _Float16* kr = Kp + (size_t)(kb + 16 * t + nl) * DK;
      v16h b0 = load_fragB(kr, hlf);
      v16h b1 = load_fragB(kr + 32, hlf);
      v8f s = {};
      s = wmma16(aQ0, b0, s);
      s = wmma16(aQ1, b1, s);
      Sv[t] = s;
    }

    // ---- causal mask (C layout: element r -> row r+8*hlf, col nl) ----
#pragma unroll
    for (int r = 0; r < 8; ++r) {
      int qg = qbase + r + 8 * hlf;
#pragma unroll
      for (int t = 0; t < 2; ++t) {
        int kg = kb + 16 * t + nl;
        if (kg > qg) Sv[t][r] = -__builtin_inff();
      }
    }

    // ---- online softmax (row reductions over the 16 lanes of each half) ----
    float alpha[8];
    v8f P0, P1;
#pragma unroll
    for (int r = 0; r < 8; ++r) {
      float rm = fmaxf(Sv[0][r], Sv[1][r]);
#pragma unroll
      for (int msk = 1; msk <= 8; msk <<= 1)
        rm = fmaxf(rm, __shfl_xor(rm, msk, 32));
      float mn = fmaxf(Mr[r], rm);
      alpha[r] = __expf(Mr[r] - mn);           // 0 on first chunk (Mr = -inf)
      Mr[r] = mn;
      float p0 = __expf(Sv[0][r] - mn);
      float p1 = __expf(Sv[1][r] - mn);
      P0[r] = p0; P1[r] = p1;
      float rs = p0 + p1;
#pragma unroll
      for (int msk = 1; msk <= 8; msk <<= 1)
        rs += __shfl_xor(rs, msk, 32);
      Lr[r] = Lr[r] * alpha[r] + rs;
    }

    // ---- P: C-layout -> LDS -> A-layout (same-wave DS ordering, no barrier) ----
#pragma unroll
    for (int r = 0; r < 8; ++r) {
      int mrow = r + 8 * hlf;
      ldsP[mrow * 32 + nl]      = (_Float16)P0[r];
      ldsP[mrow * 32 + 16 + nl] = (_Float16)P1[r];
    }
    h8 plo = *(const h8*)(&ldsP[nl * 32 + hlf * 8]);        // ds_load_b128
    h8 phi = *(const h8*)(&ldsP[nl * 32 + 16 + hlf * 8]);   // ds_load_b128
    v16h aP = __builtin_shufflevector(plo, phi,
                                      0,1,2,3,4,5,6,7,8,9,10,11,12,13,14,15);

    // ---- O = diag(alpha)*O + P.V  (V transposed: contiguous along keys) ----
#pragma unroll
    for (int t2 = 0; t2 < 4; ++t2) {
      v16h bV = load_fragB(Vp + (size_t)(16 * t2 + nl) * SEQ + kb, hlf);
#pragma unroll
      for (int r = 0; r < 8; ++r) O[t2][r] *= alpha[r];
      O[t2] = wmma16(aP, bV, O[t2]);
    }

    // prefetch next key chunk rows
    if (kb + 32 < qbase + 16)
      __builtin_prefetch(Kp + (size_t)(kb + 32 + lane) * DK, 0, 0);
  }

  // ---- epilogue: normalize, store f16 [B*S, D] for the output projection ----
  const int bb = bh >> 4;            // / NH
  const int hh = bh & (NH - 1);
#pragma unroll
  for (int t2 = 0; t2 < 4; ++t2) {
#pragma unroll
    for (int r = 0; r < 8; ++r) {
      int mrow  = r + 8 * hlf;
      int row_g = bb * SEQ + qbase + mrow;
      int col_g = hh * DK + 16 * t2 + nl;
      AO[(size_t)row_g * DMODEL + col_g] = (_Float16)(O[t2][r] / Lr[r]);
    }
  }
}

// ---------------------------------------------------------------------------
extern "C" void kernel_launch(void* const* d_in, const int* in_sizes, int n_in,
                              void* d_out, int out_size, void* d_ws, size_t ws_size,
                              hipStream_t stream)
{
  (void)in_sizes; (void)n_in; (void)out_size; (void)ws_size;

  const float* x  = (const float*)d_in[0];
  const float* wq = (const float*)d_in[1];
  const float* wk = (const float*)d_in[2];
  const float* wv = (const float*)d_in[3];
  const float* wo = (const float*)d_in[4];
  float* out = (float*)d_out;

  const size_t nX = (size_t)MROWS * DMODEL;    // 4,194,304
  const size_t nW = (size_t)DMODEL * DMODEL;   // 1,048,576

  _Float16* x16  = (_Float16*)d_ws;            // [M, D]
  _Float16* wq16 = x16 + nX;
  _Float16* wk16 = wq16 + nW;
  _Float16* wv16 = wk16 + nW;
  _Float16* wo16 = wv16 + nW;
  _Float16* Qb   = wo16 + nW;                  // [B*H, S, dk]
  _Float16* Kb   = Qb + nX;                    // [B*H, S, dk]
  _Float16* Vb   = Kb + nX;                    // [B*H, dk, S]
  _Float16* AO   = Vb + nX;                    // [M, D]
  // total workspace: 24M f16 = 48 MB

  // f32 -> f16 staging (streaming, b128 in/out)
  cvt_kernel<<<dim3((unsigned)(nX / 8 / 256)), 256, 0, stream>>>(x,  x16,  (int)nX);
  cvt_kernel<<<dim3((unsigned)(nW / 8 / 256)), 256, 0, stream>>>(wq, wq16, (int)nW);
  cvt_kernel<<<dim3((unsigned)(nW / 8 / 256)), 256, 0, stream>>>(wk, wk16, (int)nW);
  cvt_kernel<<<dim3((unsigned)(nW / 8 / 256)), 256, 0, stream>>>(wv, wv16, (int)nW);
  cvt_kernel<<<dim3((unsigned)(nW / 8 / 256)), 256, 0, stream>>>(wo, wo16, (int)nW);

  dim3 gp(MROWS / 32, DMODEL / 64);            // 128 x 16 waves, 32x64 tiles
  gemm_kernel<<<gp, 32, 0, stream>>>(x16, wq16, Qb, nullptr, 0, 0.125f); // 1/sqrt(dk)
  gemm_kernel<<<gp, 32, 0, stream>>>(x16, wk16, Kb, nullptr, 1, 1.0f);
  gemm_kernel<<<gp, 32, 0, stream>>>(x16, wv16, Vb, nullptr, 2, 1.0f);

  attn_kernel<<<dim3(NB * NH * QT), 32, 0, stream>>>(Qb, Kb, Vb, AO);

  gemm_kernel<<<gp, 32, 0, stream>>>(AO, wo16, nullptr, out, 3, 1.0f);
}
